// SubGraph_48137993453813
// MI455X (gfx1250) — compile-verified, matrix operand order checked by reading, not run on hardware
//
#include <hip/hip_runtime.h>
#include <hip/hip_bf16.h>

typedef __attribute__((ext_vector_type(16))) _Float16 v16h;
typedef __attribute__((ext_vector_type(8)))  float    v8f;
typedef __attribute__((ext_vector_type(8)))  unsigned int v8u;

#define SEQ   64     // N
#define HID   128    // H
#define CH    64     // C = H/2
#define SA    136    // half-stride for hA rows (128 K + 8 pad) -> bank-conflict free
#define SB    136    // half-stride for wB cols (128 K + 8 pad)
#define SE    68     // float-stride for encS rows (64 + 4 pad)
#define NEGV  (-10000.0f)

union HU { v8u u; v16h h; };

// Load one 16x32 (f16) WMMA operand fragment from LDS.
// linebase32: dword pointer to start of this lane's row/col K-line.
// ksDword = ks*16 (K-step in dwords), halfSel = lane>=16.
__device__ __forceinline__ v16h load_frag(const unsigned int* linebase32,
                                          int ksDword, int halfSel) {
  HU cvt;
#pragma unroll
  for (int d = 0; d < 8; ++d) {
    // dword d of operand: d<4 -> K pair {2d,2d+1}; d>=4 -> K pair {16+2(d-4),...}
    int kp = ksDword + ((d & 4) << 1) + (d & 3) + 4 * halfSel;
    cvt.u[d] = linebase32[kp];
  }
  return cvt.h;
}

__global__ __launch_bounds__(128) void fused_encoder_kernel(
    const float* __restrict__ hs,   // (B, 64, 128)
    const int*   __restrict__ li,   // (B,)
    const float* __restrict__ W,    // (3, 128, 64)
    const float* __restrict__ bb,   // (3, 64)
    const float* __restrict__ gg,   // (3, 64)
    const float* __restrict__ be,   // (3, 64)
    float* __restrict__ out)        // (B, 128)
{
  __shared__ __align__(16) _Float16 hA[SEQ * SA];   // A operand: h rows, f16
  __shared__ __align__(16) _Float16 wB[CH  * SB];   // B operand: W^T cols, f16
  __shared__ __align__(16) float    encS[SEQ * SE]; // enc (f32) for top-2 pass
  __shared__ float t1s[CH], t2s[CH], uns[CH];
  __shared__ int   idxs[CH];
  __shared__ float biasS[CH], gammaS[CH], betaS[CH];

  const int b    = blockIdx.x;
  const int tid  = threadIdx.x;
  const int lane = tid & 31;
  const int wave = tid >> 5;
  const int l16  = lane & 15;
  const int halfSel = lane >> 4;
  const int lli  = li[b];

  // ---- stage hidden_states[b] -> hA as f16 (coalesced float4 reads) ----
  {
    const float4* hs4 = (const float4*)(hs + (size_t)b * SEQ * HID);
    for (int i = tid; i < SEQ * HID / 4; i += 128) {
      int r  = i >> 5;         // 32 float4 per row
      int k4 = i & 31;
      float4 v = hs4[i];
      _Float16* dst = hA + r * SA + k4 * 4;
      dst[0] = (_Float16)v.x; dst[1] = (_Float16)v.y;
      dst[2] = (_Float16)v.z; dst[3] = (_Float16)v.w;
    }
  }

  for (int l = 0; l < 3; ++l) {
    // ---- stage W[l]^T -> wB (f16, col-major so B fragment reads are K-contiguous) ----
    const float* Wl = W + (size_t)l * HID * CH;
    for (int i = tid; i < HID * CH; i += 128) {
      int k = i >> 6, c = i & 63;            // W row-major (k, c)
      wB[c * SB + k] = (_Float16)Wl[i];
    }
    if (tid < CH) {
      biasS[tid]  = bb[l * CH + tid];
      gammaS[tid] = gg[l * CH + tid];
      betaS[tid]  = be[l * CH + tid];
    }
    __syncthreads();

    // ---- GEMM: wave owns rows [16*wave, 16*wave+16); 4 col tiles x 4 K-steps ----
    v8f acc[4];
#pragma unroll
    for (int ct = 0; ct < 4; ++ct) {
      float bv = biasS[ct * 16 + l16];       // bias broadcast per column
#pragma unroll
      for (int r = 0; r < 8; ++r) acc[ct][r] = bv;
    }
    const unsigned int* aLine =
        (const unsigned int*)(hA + (16 * wave + l16) * SA);
#pragma unroll
    for (int ks = 0; ks < 4; ++ks) {
      v16h Ak = load_frag(aLine, ks * 16, halfSel);
#pragma unroll
      for (int ct = 0; ct < 4; ++ct) {
        const unsigned int* bLine =
            (const unsigned int*)(wB + (ct * 16 + l16) * SB);
        v16h Bk = load_frag(bLine, ks * 16, halfSel);
        acc[ct] = __builtin_amdgcn_wmma_f32_16x16x32_f16(
            false, Ak, false, Bk, (short)0, acc[ct], false, false);
      }
    }

    // ---- LayerNorm epilogue (f32, in-register row reduction) ----
    float s[8], q[8];
#pragma unroll
    for (int r = 0; r < 8; ++r) {
      float a0 = acc[0][r], a1 = acc[1][r], a2 = acc[2][r], a3 = acc[3][r];
      s[r] = a0 + a1 + a2 + a3;
      q[r] = a0 * a0 + a1 * a1 + a2 * a2 + a3 * a3;
    }
#pragma unroll
    for (int m = 1; m < 16; m <<= 1) {
#pragma unroll
      for (int r = 0; r < 8; ++r) {
        s[r] += __shfl_xor(s[r], m, 32);     // stays within 16-lane half group
        q[r] += __shfl_xor(q[r], m, 32);
      }
    }
    const int rowBase = 16 * wave + 8 * halfSel;
#pragma unroll
    for (int r = 0; r < 8; ++r) {
      float mean = s[r] * (1.0f / 64.0f);
      float var  = q[r] * (1.0f / 64.0f) - mean * mean;
      float rs   = rsqrtf(var + 1e-5f);
      int row = rowBase + r;
#pragma unroll
      for (int ct = 0; ct < 4; ++ct) {
        int col = ct * 16 + l16;
        float e = (acc[ct][r] - mean) * rs * gammaS[col] + betaS[col];
        e = fmaxf(e, 0.0f);                  // relu
        encS[row * SE + col] = e;
      }
    }
    __syncthreads();

    // ---- per-column masked top-2 (+ unmasked max) over the 64 rows ----
    if (tid < CH) {
      const int c = tid;
      float t1 = -3.0e38f, t2 = -3.0e38f, un = -3.0e38f;
      int ix = 0;
      for (int n = 0; n < SEQ; ++n) {
        float e = encS[n * SE + c];
        un = fmaxf(un, e);
        float v = e + ((n >= lli) ? NEGV : 0.0f);
        if (v > t1)      { t2 = t1; t1 = v; ix = n; }
        else if (v > t2) { t2 = v; }
      }
      t1s[c] = t1; t2s[c] = t2; idxs[c] = ix; uns[c] = un;
    }
    __syncthreads();

    if (l < 2) {
      // ---- h_next = [enc | max(excl_self_max, 0)] -> hA (f16) ----
      for (int i = tid; i < SEQ * CH; i += 128) {
        int n = i >> 6, c = i & 63;
        float e  = encS[n * SE + c];
        float ex = (n == idxs[c]) ? t2s[c] : t1s[c];
        float mx = fmaxf(ex, 0.0f);
        hA[n * SA + c]      = (_Float16)e;
        hA[n * SA + 64 + c] = (_Float16)mx;
      }
      __syncthreads();
    } else {
      // ---- final: out[b, c] = max_n enc ; out[b, 64+c] = max_n mx = max(top1, 0) ----
      if (tid < CH) {
        out[(size_t)b * 128 + tid]      = uns[tid];
        out[(size_t)b * 128 + 64 + tid] = fmaxf(t1s[tid], 0.0f);
      }
    }
  }
}

extern "C" void kernel_launch(void* const* d_in, const int* in_sizes, int n_in,
                              void* d_out, int out_size, void* d_ws, size_t ws_size,
                              hipStream_t stream) {
  const float* hs = (const float*)d_in[0];
  const int*   li = (const int*)d_in[1];
  const float* W  = (const float*)d_in[2];
  const float* bb = (const float*)d_in[3];
  const float* gg = (const float*)d_in[4];
  const float* be = (const float*)d_in[5];
  float* out = (float*)d_out;

  const int B = in_sizes[1];   // li_vector_num has B elements
  dim3 grid(B), block(128);
  hipLaunchKernelGGL(fused_encoder_kernel, grid, block, 0, stream,
                     hs, li, W, bb, gg, be, out);
}
